// LJCH2_34548716929308
// MI455X (gfx1250) — matrix operand order repaired and checked by reading, not compiled, not run on hardware
//
#include <hip/hip_runtime.h>

typedef __bf16 bf16_t;
typedef __attribute__((ext_vector_type(16))) __bf16 v16bf;
typedef __attribute__((ext_vector_type(8)))  __bf16 v8bf;
typedef __attribute__((ext_vector_type(8)))  float  v8f;
typedef __attribute__((ext_vector_type(4)))  float  v4f;

namespace {
constexpr int kB = 64, kT = 512, kH = 32;
constexpr int kDc = 4096, kDd = 512, kDm = 256, kDtot = 4864;
constexpr int kRed = 128, kG = 96;        // 3*H
constexpr int kM = kB * kT;               // 32768 rows
}

// ---------------------------------------------------------------------------
// Pre-pass: transpose + convert weights to bf16 [K][N] layout for B fragments
// ---------------------------------------------------------------------------
__global__ void k_convw(const float* __restrict__ fc0_w,
                        const float* __restrict__ wih_f,
                        const float* __restrict__ wih_b,
                        bf16_t* __restrict__ Wb,
                        bf16_t* __restrict__ Wtf,
                        bf16_t* __restrict__ Wtb)
{
  const int idx = blockIdx.x * 256 + threadIdx.x;
  if (idx < kDtot * kRed) {
    const int k = idx / kRed, n = idx % kRed;
    Wb[idx] = (bf16_t)fc0_w[(size_t)n * kDtot + k];
  }
  if (idx < kRed * kG) {
    const int k = idx / kG, n = idx % kG;
    Wtf[idx] = (bf16_t)wih_f[n * kRed + k];
    Wtb[idx] = (bf16_t)wih_b[n * kRed + k];
  }
}

// ---------------------------------------------------------------------------
// GEMM1: scores[32768,128] = concat(content,distort,motion) @ fc0_w.T + b
// One wave computes a 16x128 stripe.  Output stored as bf16 row-major.
// ---------------------------------------------------------------------------
__global__ __launch_bounds__(32) void k_gemm1(
    const float* __restrict__ fc, const float* __restrict__ fd,
    const float* __restrict__ fm, const float* __restrict__ bias,
    const bf16_t* __restrict__ Wb, bf16_t* __restrict__ scoresbf)
{
  const int lane  = threadIdx.x;
  const int mrow  = lane & 15;
  const int khalf = lane >> 4;
  const int ncol  = lane & 15;
  const size_t m  = (size_t)blockIdx.x * 16 + mrow;

  const float* rowC = fc + m * (size_t)kDc;
  const float* rowD = fd + m * (size_t)kDd;
  const float* rowM = fm + m * (size_t)kDm;

  v8f acc[8];
#pragma unroll
  for (int nt = 0; nt < 8; ++nt) {
    const float bv = bias[nt * 16 + ncol];
#pragma unroll
    for (int r = 0; r < 8; ++r) acc[nt][r] = bv;
  }

  for (int k0 = 0; k0 < kDtot; k0 += 32) {       // segment boundaries are /32
    const float* src; int koff;
    if (k0 < kDc)            { src = rowC; koff = k0; }
    else if (k0 < kDc + kDd) { src = rowD; koff = k0 - kDc; }
    else                     { src = rowM; koff = k0 - kDc - kDd; }

    const float* p0 = src + koff + khalf * 8;    // 16B aligned
    v4f x0 = *(const v4f*)(p0);
    v4f x1 = *(const v4f*)(p0 + 4);
    v4f y0 = *(const v4f*)(p0 + 16);
    v4f y1 = *(const v4f*)(p0 + 20);
    v16bf a;
#pragma unroll
    for (int i = 0; i < 4; ++i) {
      a[i]      = (bf16_t)x0[i];
      a[4 + i]  = (bf16_t)x1[i];
      a[8 + i]  = (bf16_t)y0[i];
      a[12 + i] = (bf16_t)y1[i];
    }

    const bf16_t* wrow = Wb + (size_t)k0 * kRed + ncol;  // B: lane == n, elements over k (table order)
#pragma unroll
    for (int nt = 0; nt < 8; ++nt) {
      v16bf bb;
#pragma unroll
      for (int i = 0; i < 8; ++i) {
        bb[i]     = wrow[(size_t)(khalf * 8 + i) * kRed + nt * 16];
        bb[8 + i] = wrow[(size_t)(16 + khalf * 8 + i) * kRed + nt * 16];
      }
      acc[nt] = __builtin_amdgcn_wmma_f32_16x16x32_bf16(
          false, a, false, bb, (short)0, acc[nt], false, false);
    }
  }

  const size_t mb = (size_t)blockIdx.x * 16 + khalf * 8;
#pragma unroll
  for (int nt = 0; nt < 8; ++nt)
#pragma unroll
    for (int r = 0; r < 8; ++r)
      scoresbf[(mb + r) * kRed + nt * 16 + ncol] = (bf16_t)acc[nt][r];
}

// ---------------------------------------------------------------------------
// GEMM2: xg[32768,96] = scores(.reversed for bwd) @ Wih.T + bih   (f32 out)
// ---------------------------------------------------------------------------
__global__ __launch_bounds__(32) void k_gemm2(
    const bf16_t* __restrict__ scoresbf, const bf16_t* __restrict__ Wt,
    const float* __restrict__ bih, const int* __restrict__ inputLength,
    float* __restrict__ xg, int backward)
{
  const int lane  = threadIdx.x;
  const int mrow  = lane & 15;
  const int khalf = lane >> 4;
  const int ncol  = lane & 15;
  const size_t m  = (size_t)blockIdx.x * 16 + mrow;
  const int b = (int)(m >> 9);
  const int t = (int)(m & 511);
  int srow = (int)m;
  if (backward) {
    const int vlen = inputLength[b] - 9;
    const int tt = (t < vlen) ? (vlen - 1 - t) : t;
    srow = b * kT + tt;
  }
  const bf16_t* arow = scoresbf + (size_t)srow * kRed;

  v8f acc[6];
#pragma unroll
  for (int nt = 0; nt < 6; ++nt) {
    const float bv = bih[nt * 16 + ncol];
#pragma unroll
    for (int r = 0; r < 8; ++r) acc[nt][r] = bv;
  }

#pragma unroll
  for (int k0 = 0; k0 < kRed; k0 += 32) {
    v8bf lo = *(const v8bf*)(arow + k0 + khalf * 8);
    v8bf hi = *(const v8bf*)(arow + k0 + 16 + khalf * 8);
    v16bf a;
#pragma unroll
    for (int i = 0; i < 8; ++i) { a[i] = lo[i]; a[8 + i] = hi[i]; }

    const bf16_t* wrow = Wt + (size_t)k0 * kG + ncol;
#pragma unroll
    for (int nt = 0; nt < 6; ++nt) {
      v16bf bb;
#pragma unroll
      for (int i = 0; i < 8; ++i) {
        bb[i]     = wrow[(size_t)(khalf * 8 + i) * kG + nt * 16];
        bb[8 + i] = wrow[(size_t)(16 + khalf * 8 + i) * kG + nt * 16];
      }
      acc[nt] = __builtin_amdgcn_wmma_f32_16x16x32_bf16(
          false, a, false, bb, (short)0, acc[nt], false, false);
    }
  }

  const size_t mb = (size_t)blockIdx.x * 16 + khalf * 8;
#pragma unroll
  for (int nt = 0; nt < 6; ++nt)
#pragma unroll
    for (int r = 0; r < 8; ++r)
      xg[(mb + r) * (size_t)kG + nt * 16 + ncol] = acc[nt][r];
}

// ---------------------------------------------------------------------------
// GRU recurrence, f32 on VALU.  One wave per (batch, direction); lane j owns
// h[j] and the three gate rows of Whh in registers; h broadcast via LDS.
// Backward results scattered through rev() (involution) directly into out_b.
// ---------------------------------------------------------------------------
__global__ __launch_bounds__(32) void k_gru(
    const float* __restrict__ xg_f, const float* __restrict__ xg_b,
    const float* __restrict__ whh_f, const float* __restrict__ bhh_f,
    const float* __restrict__ whh_b, const float* __restrict__ bhh_b,
    const int* __restrict__ inputLength,
    float* __restrict__ out_f, float* __restrict__ out_b)
{
  __shared__ float hs[kH];
  const int b   = blockIdx.x >> 1;
  const int dir = blockIdx.x & 1;
  const int j   = threadIdx.x;

  const float* xg  = dir ? xg_b  : xg_f;
  const float* gW  = dir ? whh_b : whh_f;
  const float* gb  = dir ? bhh_b : bhh_f;
  float*       out = dir ? out_b : out_f;

  float wr[kH], wz[kH], wn[kH];
#pragma unroll
  for (int k = 0; k < kH; ++k) {
    wr[k] = gW[(0 * kH + j) * kH + k];
    wz[k] = gW[(1 * kH + j) * kH + k];
    wn[k] = gW[(2 * kH + j) * kH + k];
  }
  const float br = gb[j], bz = gb[kH + j], bn = gb[2 * kH + j];
  const int vlen = inputLength[b] - 9;

  float h = 0.0f;
  hs[j] = 0.0f;
  __syncthreads();

  for (int t = 0; t < kT; ++t) {
    const float* xr = xg + ((size_t)b * kT + t) * kG;
    float gr = br, gz = bz, gn = bn;
#pragma unroll
    for (int k = 0; k < kH; ++k) {
      const float hk = hs[k];
      gr += wr[k] * hk;
      gz += wz[k] * hk;
      gn += wn[k] * hk;
    }
    const float r  = 1.0f / (1.0f + __expf(-(xr[j] + gr)));
    const float z  = 1.0f / (1.0f + __expf(-(xr[kH + j] + gz)));
    const float n  = tanhf(xr[2 * kH + j] + r * gn);
    const float hn = (1.0f - z) * n + z * h;
    if (t < vlen) h = hn;
    const int tdst = dir ? ((t < vlen) ? (vlen - 1 - t) : t) : t;
    out[((size_t)b * kT + tdst) * kH + j] = h;
    __syncthreads();
    hs[j] = h;
    __syncthreads();
  }
}

// ---------------------------------------------------------------------------
// Tail: q projection, memory MLP + argmax, 5-tau temporal pooling, select.
// One block (512 threads == one per timestep) per batch element.
// ---------------------------------------------------------------------------
__global__ __launch_bounds__(512) void k_finalize(
    const float* __restrict__ out_f, const float* __restrict__ out_b,
    const int* __restrict__ inputLength,
    const float* __restrict__ q_w, const float* __restrict__ q_b,
    const float* __restrict__ m1w, const float* __restrict__ m1b,
    const float* __restrict__ m2w, const float* __restrict__ m2b,
    float* __restrict__ outp)
{
  __shared__ float q[kT], wv[kT], num[kT], red[kT];
  __shared__ float h1[32];
  __shared__ float s5[5];
  __shared__ int amax;

  const int b = blockIdx.x;
  const int t = threadIdx.x;
  const int vlen = inputLength[b] - 9;
  const bool valid = t < vlen;

  // q[t] = (masked outputs) . q_w + q_b
  {
    const float* fr = out_f + ((size_t)b * kT + t) * kH;
    const float* br = out_b + ((size_t)b * kT + t) * kH;
    float acc = 0.0f;
#pragma unroll
    for (int k = 0; k < kH; ++k)
      acc += fr[k] * q_w[k] + br[k] * q_w[kH + k];
    const float qt = (valid ? acc : 0.0f) + q_b[0];
    q[t]  = qt;
    const float wt = valid ? __expf(-qt) : 0.0f;   // exp(-q) * maskf
    wv[t]  = wt;
    num[t] = qt * wt;
  }
  __syncthreads();

  // memory MLP on lastHidForward / lastHidBackward
  if (t < 32) {
    const float* lf = out_f + ((size_t)b * kT + (vlen - 1)) * kH;
    const float* lb = out_b + ((size_t)b * kT + 0) * kH;
    float a1 = m1b[t];
#pragma unroll
    for (int k = 0; k < kH; ++k)
      a1 += lf[k] * m1w[t * 64 + k] + lb[k] * m1w[t * 64 + kH + k];
    h1[t] = a1 > 0.0f ? a1 : 0.0f;
  }
  __syncthreads();
  if (t == 0) {
    float best = -3.0e38f; int bi = 0;
    for (int p = 0; p < 5; ++p) {
      float lg = m2b[p];
      for (int i = 0; i < 32; ++i) lg += h1[i] * m2w[p * 32 + i];
      if (lg > best) { best = lg; bi = p; }     // softmax argmax == logit argmax
    }
    amax = bi;
  }

  const int taus[5] = {8, 10, 12, 14, 16};
  const float vlenf = (float)vlen;
  for (int ti = 0; ti < 5; ++ti) {
    const int tau = taus[ti];
    float l = q[t];                              // backward min window (inf pad)
    for (int k2 = 1; k2 < tau; ++k2) {
      const int idx = t - k2;
      if (idx >= 0) l = fminf(l, q[idx]);
    }
    float ms = 0.0f, ns = 0.0f;                  // forward sum window (0 pad)
    for (int k2 = 0; k2 < tau; ++k2) {
      const int idx = t + k2;
      if (idx < kT) { ms += num[idx]; ns += wv[idx]; }
    }
    const float ratio  = ns > 0.0f ? ms / ns : 0.0f;
    const float pooled = 0.5f * ratio + 0.5f * l;
    red[t] = valid ? pooled : 0.0f;
    __syncthreads();
    for (int s = 256; s > 0; s >>= 1) {
      if (t < s) red[t] += red[t + s];
      __syncthreads();
    }
    if (t == 0) s5[ti] = red[0] / vlenf;
    __syncthreads();
  }
  if (t == 0) outp[b] = s5[amax];
}

// ---------------------------------------------------------------------------
extern "C" void kernel_launch(void* const* d_in, const int* in_sizes, int n_in,
                              void* d_out, int out_size, void* d_ws, size_t ws_size,
                              hipStream_t stream) {
  (void)in_sizes; (void)n_in; (void)out_size; (void)ws_size;
  const float* motionFeat  = (const float*)d_in[0];
  const float* featContent = (const float*)d_in[1];
  const float* featDistort = (const float*)d_in[2];
  const int*   inputLength = (const int*)d_in[3];
  const float* fc0_w = (const float*)d_in[4];
  const float* fc0_b = (const float*)d_in[5];
  const float* wih_f = (const float*)d_in[6];
  const float* whh_f = (const float*)d_in[7];
  const float* bih_f = (const float*)d_in[8];
  const float* bhh_f = (const float*)d_in[9];
  const float* wih_b = (const float*)d_in[10];
  const float* whh_b = (const float*)d_in[11];
  const float* bih_b = (const float*)d_in[12];
  const float* bhh_b = (const float*)d_in[13];
  const float* q_w   = (const float*)d_in[14];
  const float* q_b   = (const float*)d_in[15];
  const float* m1w   = (const float*)d_in[16];
  const float* m1b   = (const float*)d_in[17];
  const float* m2w   = (const float*)d_in[18];
  const float* m2b   = (const float*)d_in[19];

  char* ws = (char*)d_ws;
  size_t off = 0;
  auto carve = [&](size_t bytes) -> char* {
    char* p = ws + off;
    off = (off + bytes + 255) & ~(size_t)255;
    return p;
  };
  bf16_t* scoresbf = (bf16_t*)carve((size_t)kM * kRed * 2);   // 8.0 MB
  bf16_t* Wb       = (bf16_t*)carve((size_t)kDtot * kRed * 2);// 1.2 MB
  bf16_t* Wtf      = (bf16_t*)carve((size_t)kRed * kG * 2);
  bf16_t* Wtb      = (bf16_t*)carve((size_t)kRed * kG * 2);
  float*  xg_f     = (float*) carve((size_t)kM * kG * 4);     // 12.6 MB
  float*  xg_b     = (float*) carve((size_t)kM * kG * 4);     // 12.6 MB
  float*  out_f    = (float*) carve((size_t)kM * kH * 4);     // 4.2 MB
  float*  out_b    = (float*) carve((size_t)kM * kH * 4);     // 4.2 MB

  k_convw<<<(kDtot * kRed + 255) / 256, 256, 0, stream>>>(
      fc0_w, wih_f, wih_b, Wb, Wtf, Wtb);
  k_gemm1<<<kM / 16, 32, 0, stream>>>(
      featContent, featDistort, motionFeat, fc0_b, Wb, scoresbf);
  k_gemm2<<<kM / 16, 32, 0, stream>>>(scoresbf, Wtf, bih_f, inputLength, xg_f, 0);
  k_gemm2<<<kM / 16, 32, 0, stream>>>(scoresbf, Wtb, bih_b, inputLength, xg_b, 1);
  k_gru<<<kB * 2, 32, 0, stream>>>(
      xg_f, xg_b, whh_f, bhh_f, whh_b, bhh_b, inputLength, out_f, out_b);
  k_finalize<<<kB, kT, 0, stream>>>(
      out_f, out_b, inputLength, q_w, q_b, m1w, m1b, m2w, m2b, (float*)d_out);
}
